// Decoder_10110353014984
// MI455X (gfx1250) — compile-verified, matrix operand order checked by reading, not run on hardware
//
#include <hip/hip_runtime.h>

typedef _Float16 h4   __attribute__((ext_vector_type(4)));
typedef _Float16 v8h  __attribute__((ext_vector_type(8)));
typedef _Float16 v16h __attribute__((ext_vector_type(16)));
typedef float    v8f  __attribute__((ext_vector_type(8)));

#define T_   4
#define B_   64
#define N_   196
#define D_   512
#define C_   1000
#define CPAD 1024
#define M_   (T_ * B_ * N_)   /* 50176 */
#define BND  (B_ * N_ * D_)   /* 6422528 */

// ---------------------------------------------------------------------------
// Kernel 1: LIF multistep scan -> binary spikes in f16 (exact {0,1}).
// ---------------------------------------------------------------------------
__global__ void __launch_bounds__(256) lif_spikes(const float* __restrict__ x,
                                                  _Float16* __restrict__ s) {
  const size_t i4 = (size_t)blockIdx.x * blockDim.x + threadIdx.x;  // over BND/4
  const float4* xp = (const float4*)x;
  h4* sp = (h4*)s;
  float4 v; v.x = 0.f; v.y = 0.f; v.z = 0.f; v.w = 0.f;
#pragma unroll
  for (int t = 0; t < T_; ++t) {
    float4 xt = xp[(size_t)t * (BND / 4) + i4];
    v.x = 0.5f * (v.x + xt.x);
    v.y = 0.5f * (v.y + xt.y);
    v.z = 0.5f * (v.z + xt.z);
    v.w = 0.5f * (v.w + xt.w);
    float s0 = (v.x >= 1.0f) ? 1.0f : 0.0f;
    float s1 = (v.y >= 1.0f) ? 1.0f : 0.0f;
    float s2 = (v.z >= 1.0f) ? 1.0f : 0.0f;
    float s3 = (v.w >= 1.0f) ? 1.0f : 0.0f;
    v.x = (s0 != 0.0f) ? 0.0f : v.x;
    v.y = (s1 != 0.0f) ? 0.0f : v.y;
    v.z = (s2 != 0.0f) ? 0.0f : v.z;
    v.w = (s3 != 0.0f) ? 0.0f : v.w;
    h4 o;
    o.x = (_Float16)s0; o.y = (_Float16)s1; o.z = (_Float16)s2; o.w = (_Float16)s3;
    sp[(size_t)t * (BND / 4) + i4] = o;
  }
}

// ---------------------------------------------------------------------------
// Kernel 2: W f32 -> f16, padded from C_=1000 to CPAD=1024 rows (zeros).
// ---------------------------------------------------------------------------
__global__ void __launch_bounds__(256) conv_w(const float* __restrict__ W,
                                              _Float16* __restrict__ Wh) {
  const int i4 = blockIdx.x * blockDim.x + threadIdx.x;  // over CPAD*D_/4
  const int c = i4 >> 7;            // /(D_/4)
  const int d4 = i4 & 127;
  h4 o = {};
  if (c < C_) {
    float4 w = ((const float4*)(W + (size_t)c * D_))[d4];
    o.x = (_Float16)w.x; o.y = (_Float16)w.y;
    o.z = (_Float16)w.z; o.w = (_Float16)w.w;
  }
  ((h4*)Wh)[i4] = o;
}

// ---------------------------------------------------------------------------
// WMMA helpers
// ---------------------------------------------------------------------------
__device__ __forceinline__ v16h ldfrag(const _Float16* p) {
  // ISA 16-bit fragment layout: per lane, two contiguous 16B chunks
  // (k+{0..7} and k+{16..23} for lanes 0-15; +8 shifted for lanes 16-31 —
  // the caller bakes the +8 into p via koff).
  v8h lo = *(const v8h*)(p);
  v8h hi = *(const v8h*)(p + 16);
  return __builtin_shufflevector(lo, hi,
      0, 1, 2, 3, 4, 5, 6, 7, 8, 9, 10, 11, 12, 13, 14, 15);
}

__device__ __forceinline__ v8f wmma(v16h a, v16h b, v8f c) {
  return __builtin_amdgcn_wmma_f32_16x16x32_f16(false, a, false, b,
                                                (short)0, c, false, false);
}

__device__ __forceinline__ void store_tile(float* __restrict__ out,
                                           const float* __restrict__ bias,
                                           int row, int col, v8f a) {
  if (col < C_) {
    const float bb = bias[col];
#pragma unroll
    for (int r = 0; r < 8; ++r) {
      out[(size_t)(row + r) * C_ + col] = a[r] + bb;
    }
  }
}

// ---------------------------------------------------------------------------
// Kernel 3: WMMA GEMM. out[m][c] = sum_k A[m][k] * Wh[c][k] + bias[c]
//   A: [M_][D_] f16 spikes, Wh: [CPAD][D_] f16, out: [M_][C_] f32.
// Each wave: 32 rows x 64 cols (2 A frags x 4 B frags -> 8 WMMAs / k-step),
// software-pipelined: next k-step's 12 fragment loads are issued before the
// current k-step's WMMAs so the scheduler overlaps loads with matrix math.
// Block = 8 waves = 256 rows. Grid = (M_/256, CPAD/64).
// ---------------------------------------------------------------------------
__global__ void __launch_bounds__(256) spike_gemm(const _Float16* __restrict__ A,
                                                  const _Float16* __restrict__ Bw,
                                                  const float* __restrict__ bias,
                                                  float* __restrict__ out) {
  const int lane = threadIdx.x & 31;
  const int wave = threadIdx.x >> 5;
  const int mr   = lane & 15;     // row within tile (A) / col within tile (B)
  const int hi   = lane >> 4;     // K-chunk half select
  const int koff = hi << 3;       // 0 or 8

  const int m0 = blockIdx.x * 256 + wave * 32;
  const int n0 = blockIdx.y * 64;

  const _Float16* ap0 = A + (size_t)(m0 + mr) * D_ + koff;
  const _Float16* ap1 = ap0 + 16 * D_;
  const _Float16* bp0 = Bw + (size_t)(n0 + mr) * D_ + koff;
  const _Float16* bp1 = bp0 + 16 * D_;
  const _Float16* bp2 = bp0 + 32 * D_;
  const _Float16* bp3 = bp0 + 48 * D_;

  v8f c00 = {}, c01 = {}, c02 = {}, c03 = {};
  v8f c10 = {}, c11 = {}, c12 = {}, c13 = {};

  // Prologue: fragments for k = 0.
  v16h a0 = ldfrag(ap0);
  v16h a1 = ldfrag(ap1);
  v16h b0 = ldfrag(bp0);
  v16h b1 = ldfrag(bp1);
  v16h b2 = ldfrag(bp2);
  v16h b3 = ldfrag(bp3);

  for (int k0 = 32; k0 < D_; k0 += 32) {
    // Issue next k-step's loads first (pipelining).
    v16h na0 = ldfrag(ap0 + k0);
    v16h na1 = ldfrag(ap1 + k0);
    v16h nb0 = ldfrag(bp0 + k0);
    v16h nb1 = ldfrag(bp1 + k0);
    v16h nb2 = ldfrag(bp2 + k0);
    v16h nb3 = ldfrag(bp3 + k0);

    c00 = wmma(a0, b0, c00);
    c01 = wmma(a0, b1, c01);
    c02 = wmma(a0, b2, c02);
    c03 = wmma(a0, b3, c03);
    c10 = wmma(a1, b0, c10);
    c11 = wmma(a1, b1, c11);
    c12 = wmma(a1, b2, c12);
    c13 = wmma(a1, b3, c13);

    a0 = na0; a1 = na1;
    b0 = nb0; b1 = nb1; b2 = nb2; b3 = nb3;
  }

  // Tail k-step.
  c00 = wmma(a0, b0, c00);
  c01 = wmma(a0, b1, c01);
  c02 = wmma(a0, b2, c02);
  c03 = wmma(a0, b3, c03);
  c10 = wmma(a1, b0, c10);
  c11 = wmma(a1, b1, c11);
  c12 = wmma(a1, b2, c12);
  c13 = wmma(a1, b3, c13);

  // Epilogue: C/D layout — lanes 0-15: rows +0..7, col = lane;
  // lanes 16-31: rows +8..15, col = lane-16. Add bias, skip padded cols.
  const int rowb = m0 + (hi << 3);
  store_tile(out, bias, rowb,      n0 +  0 + mr, c00);
  store_tile(out, bias, rowb,      n0 + 16 + mr, c01);
  store_tile(out, bias, rowb,      n0 + 32 + mr, c02);
  store_tile(out, bias, rowb,      n0 + 48 + mr, c03);
  store_tile(out, bias, rowb + 16, n0 +  0 + mr, c10);
  store_tile(out, bias, rowb + 16, n0 + 16 + mr, c11);
  store_tile(out, bias, rowb + 16, n0 + 32 + mr, c12);
  store_tile(out, bias, rowb + 16, n0 + 48 + mr, c13);
}

extern "C" void kernel_launch(void* const* d_in, const int* in_sizes, int n_in,
                              void* d_out, int out_size, void* d_ws, size_t ws_size,
                              hipStream_t stream) {
  (void)in_sizes; (void)n_in; (void)out_size; (void)ws_size;
  const float* x    = (const float*)d_in[0];  // [T,B,N,D] f32
  const float* W    = (const float*)d_in[1];  // [C,D] f32
  const float* bias = (const float*)d_in[2];  // [C] f32
  float* out = (float*)d_out;                 // [T,B,N,C] f32

  _Float16* spikes = (_Float16*)d_ws;                              // M_*D_ f16
  _Float16* Wh = (_Float16*)((char*)d_ws + (size_t)M_ * D_ * 2);   // CPAD*D_ f16

  // 1) LIF scan -> spikes (f16)
  lif_spikes<<<(BND / 4) / 256, 256, 0, stream>>>(x, spikes);
  // 2) W -> f16, padded to CPAD rows
  conv_w<<<(CPAD * D_ / 4) / 256, 256, 0, stream>>>(W, Wh);
  // 3) WMMA GEMM + bias
  dim3 grid(M_ / 256, CPAD / 64);
  spike_gemm<<<grid, 256, 0, stream>>>(spikes, Wh, bias, out);
}
